// MultiHeadAttention_33586644255005
// MI455X (gfx1250) — compile-verified, matrix operand order checked by reading, not run on hardware
//
#include <hip/hip_runtime.h>
#include <cstdint>

// Problem dims (fixed by reference)
#define B_   2
#define S_   2048
#define D_   2048
#define H_   16
#define HD_  128
#define BS_  (B_ * S_)        // 4096
#define BSD_ (BS_ * D_)       // 8388608
#define DD_  (D_ * D_)        // 4194304

typedef unsigned short bf16_t;                       // bf16 storage (host-safe)
typedef __bf16 v16bf __attribute__((ext_vector_type(16)));
typedef float  v8f   __attribute__((ext_vector_type(8)));
typedef unsigned int u32x4 __attribute__((ext_vector_type(4)));
typedef int  i32x4 __attribute__((ext_vector_type(4)));
typedef int  i32x8 __attribute__((ext_vector_type(8)));

#if defined(__HIP_DEVICE_COMPILE__) && __has_builtin(__builtin_amdgcn_tensor_load_to_lds)
#define HAVE_TDM 1
#else
#define HAVE_TDM 0
#endif

__device__ __forceinline__ bf16_t f32_to_bf16(float f) {
  union { float f; uint32_t u; } a; a.f = f;
  uint32_t r = a.u + 0x7FFFu + ((a.u >> 16) & 1u);   // round-to-nearest-even
  return (bf16_t)(r >> 16);
}

__device__ __forceinline__ v8f vzero8() {
  v8f z;
#pragma unroll
  for (int i = 0; i < 8; ++i) z[i] = 0.0f;
  return z;
}

// ---- CDNA5 async copy: global -> LDS, 16 bytes per lane, tracked by ASYNCcnt.
__device__ __forceinline__ void async_copy16(uint32_t lds_addr, const bf16_t* gptr) {
  asm volatile("global_load_async_to_lds_b128 %0, %1, off"
               :: "v"(lds_addr), "v"((uint64_t)(uintptr_t)gptr)
               : "memory");
}
__device__ __forceinline__ void wait_async0() {
  asm volatile("s_wait_asynccnt 0" ::: "memory");
}
__device__ __forceinline__ void wait_tensor0() {
#if defined(__HIP_DEVICE_COMPILE__) && __has_builtin(__builtin_amdgcn_s_wait_tensorcnt)
  __builtin_amdgcn_s_wait_tensorcnt(0);
#else
  asm volatile("s_wait_tensorcnt 0x0" ::: "memory");
#endif
}
__device__ __forceinline__ uint32_t lds_off(const void* p) {
  return (uint32_t)(uintptr_t)p;   // LDS aperture: offset in addr[31:0]
}

// ---- Tensor Data Mover: 2D tile (dim0 = contiguous x, dim1 = lines) of 2-byte
// elements, global -> LDS (tile rows packed contiguously in LDS). D# layout per
// CDNA5 ISA ch.8: group0 {count=1, lds_addr, global_addr, type=2}, group1
// {data_size=1(2B), tensor_dim0/1, tile_dim0/1, tensor_dim0_stride}.
__device__ __forceinline__ void tdm_load_2d(uint32_t lds_addr, const void* gaddr,
                                            uint32_t dim0, uint32_t dim1,
                                            uint32_t tile0, uint32_t tile1,
                                            uint32_t stride0_elems) {
#if HAVE_TDM
  uint64_t ga = (uint64_t)(uintptr_t)gaddr;
  u32x4 g0;
  g0[0] = 1u;                                              // count=1 (valid), user
  g0[1] = lds_addr;                                        // LDS byte address
  g0[2] = (uint32_t)ga;                                    // global_addr[31:0]
  g0[3] = (uint32_t)((ga >> 32) & 0x01FFFFFFu) | 0x80000000u; // addr[56:32] | type=2
  i32x8 g1;
  g1[0] = 0x00010000;                                      // data_size=1 (2 bytes)
  g1[1] = (int)((dim0 & 0xFFFFu) << 16);                   // tensor_dim0[15:0]
  g1[2] = (int)(((dim0 >> 16) & 0xFFFFu) | ((dim1 & 0xFFFFu) << 16));
  g1[3] = (int)(((dim1 >> 16) & 0xFFFFu) | ((tile0 & 0xFFFFu) << 16));
  g1[4] = (int)(tile1 & 0xFFFFu);                          // tile_dim1 ; tile_dim2=0
  g1[5] = (int)stride0_elems;                              // tensor_dim0_stride lo32
  g1[6] = 0;                                               // stride hi / dim1_stride
  g1[7] = 0;
  i32x4 z4 = {0, 0, 0, 0};
#if __clang_major__ >= 23
  i32x8 z8 = {0, 0, 0, 0, 0, 0, 0, 0};
  __builtin_amdgcn_tensor_load_to_lds(g0, g1, z4, z4, z8, 0);
#else
  __builtin_amdgcn_tensor_load_to_lds(g0, g1, z4, z4, 0);
#endif
#else
  (void)lds_addr; (void)gaddr; (void)dim0; (void)dim1;
  (void)tile0; (void)tile1; (void)stride0_elems;
#endif
}

// Load a 16x32 bf16 WMMA fragment (A or B operand). `base` points at element
// [tile_row0, k0] of a K-contiguous row-major matrix with leading dim `ld`.
__device__ __forceinline__ v16bf load_frag(const bf16_t* base, int ld) {
  const int lane = threadIdx.x & 31;
  const bf16_t* p = base + (lane & 15) * ld + ((lane >> 4) << 4);
  return *(const v16bf*)p;
}

__device__ __forceinline__ v8f wmma_bf16(v16bf a, v16bf b, v8f c) {
  return __builtin_amdgcn_wmma_f32_16x16x32_bf16(
      /*neg_a=*/false, a, /*neg_b=*/false, b,
      /*c_mod=*/(short)0, c, /*reuse_a=*/false, /*reuse_b=*/false);
}

// ---------------------------------------------------------------------------
// f32 -> bf16 conversion
// ---------------------------------------------------------------------------
__global__ __launch_bounds__(256) void cvt_f32_bf16(const float* __restrict__ src,
                                                    bf16_t* __restrict__ dst, int n) {
  int i = blockIdx.x * blockDim.x + threadIdx.x;
  if (i < n) dst[i] = f32_to_bf16(src[i]);
}

// ---------------------------------------------------------------------------
// C[M,N](f32) = A[M,K](bf16, K-contig) * Bt[N,K](bf16, K-contig)^T
// Block tile 128x256, 8 waves (2x4), wave tile 64x64 (16 WMMAs / K-step).
// Double-buffered LDS fed by global_load_async_to_lds_b128 (ASYNCcnt).
// ---------------------------------------------------------------------------
__global__ __launch_bounds__(256) void gemm_bf16_kernel(
    const bf16_t* __restrict__ A, const bf16_t* __restrict__ Bt,
    float* __restrict__ C, int M, int N, int K) {
  __shared__ bf16_t As[2][128 * 32];   // 2 x 8 KB
  __shared__ bf16_t Bs[2][256 * 32];   // 2 x 16 KB

  const int tid  = threadIdx.x;
  const int lane = tid & 31;
  const int wave = tid >> 5;
  const int wm   = wave >> 2;          // 0..1 (M)
  const int wn   = wave & 3;           // 0..3 (N)
  const int bm   = blockIdx.y * 128;
  const int bn   = blockIdx.x * 256;

  v8f acc[4][4];
#pragma unroll
  for (int mi = 0; mi < 4; ++mi)
#pragma unroll
    for (int ni = 0; ni < 4; ++ni) acc[mi][ni] = vzero8();

  // Copy slices: A -> thread t covers 32 B of row t>>1; B -> 64 B of row t.
  const int ra  = tid >> 1;
  const int kha = (tid & 1) * 16;
  const bf16_t* aRow = A  + (size_t)(bm + ra) * K + kha;
  const bf16_t* bRow = Bt + (size_t)(bn + tid) * K;
  const uint32_t laA = lds_off(&As[0][ra * 32 + kha]);
  const uint32_t laB = lds_off(&Bs[0][tid * 32]);
  const uint32_t bufA = 128 * 32 * (uint32_t)sizeof(bf16_t);   // 8192 B
  const uint32_t bufB = 256 * 32 * (uint32_t)sizeof(bf16_t);   // 16384 B

  auto issue = [&](int bufi, int k0) {
    uint32_t la = laA + (uint32_t)bufi * bufA;
    uint32_t lb = laB + (uint32_t)bufi * bufB;
    async_copy16(la,      aRow + k0);
    async_copy16(la + 16, aRow + k0 + 8);
    async_copy16(lb,      bRow + k0);
    async_copy16(lb + 16, bRow + k0 + 8);
    async_copy16(lb + 32, bRow + k0 + 16);
    async_copy16(lb + 48, bRow + k0 + 24);
  };

  issue(0, 0);                          // prologue: stage first slab

  for (int k0 = 0; k0 < K; k0 += 32) {
    const int cur = (k0 >> 5) & 1;
    wait_async0();                      // this wave's copies for `cur` landed
    __syncthreads();                    // everyone's copies landed; prev compute done
    if (k0 + 32 < K) issue(cur ^ 1, k0 + 32);        // overlap next slab copy
    if (k0 + 64 < K) {                  // warm L2 for slab after next
      __builtin_prefetch(aRow + k0 + 64, 0, 1);
      __builtin_prefetch(bRow + k0 + 64, 0, 1);
    }

    const bf16_t* as = As[cur];
    const bf16_t* bs = Bs[cur];
    v16bf af[4], bfr[4];
#pragma unroll
    for (int mi = 0; mi < 4; ++mi)
      af[mi] = load_frag(&as[(wm * 64 + mi * 16) * 32], 32);
#pragma unroll
    for (int ni = 0; ni < 4; ++ni)
      bfr[ni] = load_frag(&bs[(wn * 64 + ni * 16) * 32], 32);
#pragma unroll
    for (int mi = 0; mi < 4; ++mi)
#pragma unroll
      for (int ni = 0; ni < 4; ++ni)
        acc[mi][ni] = wmma_bf16(af[mi], bfr[ni], acc[mi][ni]);
  }

  // Epilogue: C-frag layout is VGPR j -> M=j+8*(lane>=16), lane&15 -> N.
#pragma unroll
  for (int mi = 0; mi < 4; ++mi)
#pragma unroll
    for (int ni = 0; ni < 4; ++ni)
#pragma unroll
      for (int j = 0; j < 8; ++j) {
        int row = bm + wm * 64 + mi * 16 + j + ((lane >> 4) << 3);
        int col = bn + wn * 64 + ni * 16 + (lane & 15);
        C[(size_t)row * N + col] = acc[mi][ni][j];
      }
}

// ---------------------------------------------------------------------------
// RoPE on Q,K (f32 in) + head-split pack to bf16:
//   qb, kb : [B,H,S,HD]   (K-contig over HD for QK^T WMMA)
//   vtb    : [B,H,HD,S]   (K-contig over S for PV WMMA — V pre-transposed)
// ---------------------------------------------------------------------------
__global__ __launch_bounds__(256) void rope_pack_kernel(
    const float* __restrict__ qf, const float* __restrict__ kf,
    const float* __restrict__ vf, const float* __restrict__ cosT,
    const float* __restrict__ sinT, bf16_t* __restrict__ qb,
    bf16_t* __restrict__ kb, bf16_t* __restrict__ vtb) {
  int i = blockIdx.x * blockDim.x + threadIdx.x;
  if (i >= BSD_) return;
  int col = i % D_;
  int row = i / D_;          // b*S + s
  int s   = row % S_;
  int b   = row / S_;
  int h   = col / HD_;
  int d   = col % HD_;

  float c  = cosT[s * HD_ + d];
  float sn = sinT[s * HD_ + d];
  int   partner = (d < HD_ / 2) ? (col + HD_ / 2) : (col - HD_ / 2);

  float q1 = qf[i], q2 = qf[(size_t)row * D_ + partner];
  float k1 = kf[i], k2 = kf[(size_t)row * D_ + partner];
  float qrot = (d < HD_ / 2) ? -q2 : q2;   // rotated = [-x2, x1]
  float krot = (d < HD_ / 2) ? -k2 : k2;

  size_t hidx = ((size_t)(b * H_ + h) * S_ + s) * HD_ + d;
  qb[hidx] = f32_to_bf16(q1 * c + qrot * sn);
  kb[hidx] = f32_to_bf16(k1 * c + krot * sn);
  vtb[((size_t)(b * H_ + h) * HD_ + d) * S_ + s] = f32_to_bf16(vf[i]);
}

// ---------------------------------------------------------------------------
// Flash attention (causal). 1 block = (b, h, 64-row q tile), 4 waves,
// each wave owns 16 q rows. K/Vt tiles staged to LDS by the Tensor Data
// Mover (wave 0: K tile D#, wave 1: Vt tile D#, TENSORcnt), online softmax
// in the C-frag register layout, P re-laid out via LDS, 32 WMMAs per tile.
// ---------------------------------------------------------------------------
__global__ __launch_bounds__(128) void attn_kernel(
    const bf16_t* __restrict__ qb, const bf16_t* __restrict__ kb,
    const bf16_t* __restrict__ vtb, bf16_t* __restrict__ ctxb) {
  __shared__ bf16_t Ks[64 * HD_];      // K tile  [s_k][hd]   16 KB
  __shared__ bf16_t Vts[HD_ * 64];     // Vt tile [hd][s_k]   16 KB
  __shared__ bf16_t Ps[4 * 16 * 64];   // per-wave P tiles     8 KB

  const int tid  = threadIdx.x;
  const int lane = tid & 31;
  const int wave = tid >> 5;

  const int nqt = S_ / 64;
  const int qt  = blockIdx.x % nqt;
  const int hh  = (blockIdx.x / nqt) % H_;
  const int bb  = blockIdx.x / (nqt * H_);

  const size_t headq = (size_t)(bb * H_ + hh) * S_;
  const size_t headv = (size_t)(bb * H_ + hh) * HD_;

  // Q fragments for this wave's 16 rows, all 128 K(=HD) in registers.
  v16bf qfrag[4];
#pragma unroll
  for (int kk = 0; kk < 4; ++kk)
    qfrag[kk] = load_frag(qb + (headq + qt * 64 + wave * 16) * HD_ + kk * 32, HD_);

  v8f oacc[8];
#pragma unroll
  for (int i = 0; i < 8; ++i) oacc[i] = vzero8();
  float mrow[8], lrow[8];
#pragma unroll
  for (int j = 0; j < 8; ++j) { mrow[j] = -1e30f; lrow[j] = 0.0f; }

  const float sscale    = 0.088388347648318447f;  // 1/sqrt(128)
  const int   qrow_base = qt * 64 + wave * 16 + ((lane >> 4) << 3);

#if !HAVE_TDM
  // Fallback async-copy slices (fixed across iterations)
  const int rK = tid >> 1, segK = (tid & 1) * 64;
  const uint32_t lK = lds_off(&Ks[rK * HD_ + segK]);
  const uint32_t lV = lds_off(&Vts[tid * 64]);
#endif

  for (int kt = 0; kt <= qt; ++kt) {
    __syncthreads();   // previous iteration's reads of Ks/Vts complete
#if HAVE_TDM
    if (wave == 0) {           // K tile: 64 lines x 128 elems, line stride HD_
      tdm_load_2d(lds_off(Ks), kb + (headq + (size_t)kt * 64) * HD_,
                  /*dim0=*/HD_, /*dim1=*/64, /*tile0=*/HD_, /*tile1=*/64,
                  /*stride0=*/HD_);
      wait_tensor0();
    } else if (wave == 1) {    // Vt tile: 128 lines x 64 elems, line stride S_
      tdm_load_2d(lds_off(Vts), vtb + headv * S_ + (size_t)kt * 64,
                  /*dim0=*/64, /*dim1=*/HD_, /*tile0=*/64, /*tile1=*/HD_,
                  /*stride0=*/S_);
      wait_tensor0();
    } else if (kt < qt) {      // waves 2..3: warm L2 for next key tile
      int idx = (wave - 2) * 32 + lane;   // 0..63
      __builtin_prefetch(kb + (headq + (size_t)(kt + 1) * 64 + idx) * HD_, 0, 1);
      __builtin_prefetch(vtb + (headv + idx * 2) * S_ + (kt + 1) * 64, 0, 1);
    }
#else
    {
      const bf16_t* gk = kb + (headq + (size_t)kt * 64 + rK) * HD_ + segK;
      const bf16_t* gv = vtb + (headv + tid) * S_ + kt * 64;
#pragma unroll
      for (int c = 0; c < 8; ++c) async_copy16(lK + c * 16, gk + c * 8);
#pragma unroll
      for (int c = 0; c < 8; ++c) async_copy16(lV + c * 16, gv + c * 8);
    }
    wait_async0();
#endif
    __syncthreads();   // all tile data landed in LDS

    // S = Q * K^T : 16 x 64 per wave
    v8f sacc[4];
#pragma unroll
    for (int n = 0; n < 4; ++n) sacc[n] = vzero8();
#pragma unroll
    for (int kk = 0; kk < 4; ++kk)
#pragma unroll
      for (int n = 0; n < 4; ++n) {
        v16bf kfrag = load_frag(&Ks[(n * 16) * HD_ + kk * 32], HD_);
        sacc[n] = wmma_bf16(qfrag[kk], kfrag, sacc[n]);
      }

    // scale + causal mask
    const int kcol = kt * 64 + (lane & 15);
#pragma unroll
    for (int n = 0; n < 4; ++n)
#pragma unroll
      for (int j = 0; j < 8; ++j) {
        float s = sacc[n][j] * sscale;
        if (kcol + n * 16 > qrow_base + j) s = -1e30f;
        sacc[n][j] = s;
      }

    // online softmax per row (rows live in 16-lane half-wave groups)
    bf16_t* pw = &Ps[wave * 16 * 64];
#pragma unroll
    for (int j = 0; j < 8; ++j) {
      float rmax = fmaxf(fmaxf(sacc[0][j], sacc[1][j]), fmaxf(sacc[2][j], sacc[3][j]));
#pragma unroll
      for (int m = 1; m < 16; m <<= 1) rmax = fmaxf(rmax, __shfl_xor(rmax, m, 32));
      float mnew = fmaxf(mrow[j], rmax);
      float corr = __expf(mrow[j] - mnew);
      float rsum = 0.0f;
#pragma unroll
      for (int n = 0; n < 4; ++n) {
        float p = __expf(sacc[n][j] - mnew);
        sacc[n][j] = p;
        rsum += p;
      }
#pragma unroll
      for (int m = 1; m < 16; m <<= 1) rsum += __shfl_xor(rsum, m, 32);
      lrow[j] = lrow[j] * corr + rsum;
      mrow[j] = mnew;
#pragma unroll
      for (int nf = 0; nf < 8; ++nf) oacc[nf][j] *= corr;   // rescale O row
    }

    // P (C-frag layout) -> LDS as row-major 16x64 bf16, then reload as A-frags.
#pragma unroll
    for (int n = 0; n < 4; ++n)
#pragma unroll
      for (int j = 0; j < 8; ++j)
        pw[(j + ((lane >> 4) << 3)) * 64 + n * 16 + (lane & 15)] =
            f32_to_bf16(sacc[n][j]);

    // O += P * V  (k-dim = 64 keys, 2 WMMA K-steps x 8 hd fragments)
#pragma unroll
    for (int ks = 0; ks < 2; ++ks) {
      v16bf pfrag = load_frag(&pw[ks * 32], 64);
#pragma unroll
      for (int nf = 0; nf < 8; ++nf) {
        v16bf vfrag = load_frag(&Vts[(nf * 16) * 64 + ks * 32], 64);
        oacc[nf] = wmma_bf16(pfrag, vfrag, oacc[nf]);
      }
    }
  }

  // normalize + store merged-head context (bf16, row-major [B*S, D])
#pragma unroll
  for (int j = 0; j < 8; ++j) {
    float inv = 1.0f / lrow[j];
    int q = qt * 64 + wave * 16 + ((lane >> 4) << 3) + j;
#pragma unroll
    for (int nf = 0; nf < 8; ++nf) {
      int col = hh * HD_ + nf * 16 + (lane & 15);
      ctxb[((size_t)(bb * S_ + q)) * D_ + col] = f32_to_bf16(oacc[nf][j] * inv);
    }
  }
}

// ---------------------------------------------------------------------------
// Host-side orchestration
// ---------------------------------------------------------------------------
extern "C" void kernel_launch(void* const* d_in, const int* in_sizes, int n_in,
                              void* d_out, int out_size, void* d_ws, size_t ws_size,
                              hipStream_t stream) {
  (void)in_sizes; (void)n_in; (void)out_size; (void)ws_size;
  const float* x    = (const float*)d_in[0];
  const float* Wq   = (const float*)d_in[1];
  const float* Wk   = (const float*)d_in[2];
  const float* Wv   = (const float*)d_in[3];
  const float* Wo   = (const float*)d_in[4];
  const float* cosT = (const float*)d_in[5];
  const float* sinT = (const float*)d_in[6];

  char* w = (char*)d_ws;
  bf16_t* xb   = (bf16_t*)w; w += (size_t)BSD_ * 2;
  bf16_t* wqb  = (bf16_t*)w; w += (size_t)DD_ * 2;
  bf16_t* wkb  = (bf16_t*)w; w += (size_t)DD_ * 2;
  bf16_t* wvb  = (bf16_t*)w; w += (size_t)DD_ * 2;
  bf16_t* wob  = (bf16_t*)w; w += (size_t)DD_ * 2;
  float*  qf   = (float*)w;  w += (size_t)BSD_ * 4;
  float*  kf   = (float*)w;  w += (size_t)BSD_ * 4;
  float*  vf   = (float*)w;  w += (size_t)BSD_ * 4;
  bf16_t* qb   = (bf16_t*)w; w += (size_t)BSD_ * 2;
  bf16_t* kb   = (bf16_t*)w; w += (size_t)BSD_ * 2;
  bf16_t* vtb  = (bf16_t*)w; w += (size_t)BSD_ * 2;
  bf16_t* ctxb = (bf16_t*)w; w += (size_t)BSD_ * 2;

  cvt_f32_bf16<<<(BSD_ + 255) / 256, 256, 0, stream>>>(x,  xb,  BSD_);
  cvt_f32_bf16<<<(DD_  + 255) / 256, 256, 0, stream>>>(Wq, wqb, DD_);
  cvt_f32_bf16<<<(DD_  + 255) / 256, 256, 0, stream>>>(Wk, wkb, DD_);
  cvt_f32_bf16<<<(DD_  + 255) / 256, 256, 0, stream>>>(Wv, wvb, DD_);
  cvt_f32_bf16<<<(DD_  + 255) / 256, 256, 0, stream>>>(Wo, wob, DD_);

  dim3 gg(D_ / 256, BS_ / 128);   // (8, 32)
  gemm_bf16_kernel<<<gg, 256, 0, stream>>>(xb, wqb, qf, BS_, D_, D_);
  gemm_bf16_kernel<<<gg, 256, 0, stream>>>(xb, wkb, kf, BS_, D_, D_);
  gemm_bf16_kernel<<<gg, 256, 0, stream>>>(xb, wvb, vf, BS_, D_, D_);

  rope_pack_kernel<<<(BSD_ + 255) / 256, 256, 0, stream>>>(
      qf, kf, vf, cosT, sinT, qb, kb, vtb);

  attn_kernel<<<B_ * H_ * (S_ / 64), 128, 0, stream>>>(qb, kb, vtb, ctxb);

  gemm_bf16_kernel<<<gg, 256, 0, stream>>>(ctxb, wob, (float*)d_out, BS_, D_, D_);
}
